// PointerLayer_79113297592388
// MI455X (gfx1250) — matrix operand order repaired and checked by reading, not compiled, not run on hardware
//
#include <hip/hip_runtime.h>
#include <hip/hip_bf16.h>

// ---------------- problem constants ----------------
constexpr int B_ = 2, N_ = 2048, D_ = 1024, H_ = 16, TOPK_ = 8, HIST_ = 8;
constexpr int DH_ = D_ / H_;          // 64
constexpr int DFF_ = 2730;
constexpr int DFFP_ = 2752;           // padded: 43*64 = 86*32
constexpr int M_ = B_ * N_;           // 4096 tokens
constexpr float EPS_ = 1e-6f;

typedef _Float16 f16;
typedef _Float16 v16h __attribute__((ext_vector_type(16)));
typedef _Float16 v8h  __attribute__((ext_vector_type(8)));
typedef float    v8f  __attribute__((ext_vector_type(8)));

// ---- CDNA5 async copy helpers (CDNA5 ISA) ----
__device__ __forceinline__ void async_wait0() {
  asm volatile("s_wait_asynccnt 0" ::: "memory");
}

// copy one 16B segment global -> LDS, tracked by ASYNCcnt
__device__ __forceinline__ void async_g2l_b128(uint32_t lds_addr, uint64_t gaddr) {
  asm volatile("global_load_async_to_lds_b128 %0, %1, off"
               :: "v"(lds_addr), "v"(gaddr) : "memory");
}

// =====================================================================
// WMMA GEMM:  C[M x Nout](f16) = A[M x K](f16,row-major) * W[Nout x K]^T
// Block = 4 waves = 64(M) x 64(N) tile. Weight chunk (64 rows x 32 K)
// is double-buffered in LDS via global_load_async_to_lds_b128; B
// fragments are read back with ds loads; A fragments stream from
// global. v_wmma_f32_16x16x32_f16 with f32 accumulate.
// Fragment layouts per CDNA5 ISA 7.12.2.
// =====================================================================
__global__ void __launch_bounds__(128)
wmma_gemm_xwT(const f16* __restrict__ A, const f16* __restrict__ W,
              f16* __restrict__ C, int K, int lda, int ldb, int ldc) {
  __shared__ f16 bbuf[2][64 * 32];     // 2 x 4KB double buffer

  const int wave = threadIdx.x >> 5;
  const int lane = threadIdx.x & 31;
  const int l15  = lane & 15;
  const int hi8  = (lane >> 4) << 3;   // 0 for lanes 0-15, 8 for lanes 16-31
  const int m0   = blockIdx.x * 64 + wave * 16;
  const int n0   = blockIdx.y << 6;

  v8f acc[4];
#pragma unroll
  for (int t = 0; t < 4; ++t)
#pragma unroll
    for (int r = 0; r < 8; ++r) acc[t][r] = 0.f;

  const f16* arow = A + (size_t)(m0 + l15) * lda;   // A row = lane&15

  const uint32_t lds0 = (uint32_t)(uintptr_t)(&bbuf[0][0]);
  const uint32_t lds1 = (uint32_t)(uintptr_t)(&bbuf[1][0]);

  // issue one 64x32 weight chunk: 256 x 16B segments, 2 per thread
  auto issue_chunk = [&](int k0, uint32_t lbase) {
#pragma unroll
    for (int j = 0; j < 2; ++j) {
      const int s   = wave * 64 + j * 32 + lane;    // segment 0..255
      const int row = s >> 2, seg = s & 3;          // 4 x 16B per weight row
      uint64_t ga = (uint64_t)(uintptr_t)(W + (size_t)(n0 + row) * ldb + k0)
                    + (uint64_t)(seg * 16);
      async_g2l_b128(lbase + (uint32_t)(s * 16), ga);
    }
  };

  // prologue: chunk 0 in flight
  issue_chunk(0, lds0);
  async_wait0();
  __syncthreads();

  const int nch = K >> 5;
  for (int i = 0; i < nch; ++i) {
    const int k0 = i << 5;
    const f16* bcur = &bbuf[i & 1][0];

    if (i + 1 < nch) {
      issue_chunk(k0 + 32, (i & 1) ? lds0 : lds1);
      __builtin_prefetch(arow + k0 + 32, 0, 1);     // global_prefetch next A chunk
    }

    // A fragment: halves 0..7 -> K=k0+hi8.. ; halves 8..15 -> K=k0+16+hi8..
    v8h alo = *(const v8h*)(arow + k0 + hi8);
    v8h ahi = *(const v8h*)(arow + k0 + 16 + hi8);
    v16h a;
#pragma unroll
    for (int q = 0; q < 8; ++q) { a[q] = alo[q]; a[q + 8] = ahi[q]; }

#pragma unroll
    for (int t = 0; t < 4; ++t) {
      // B fragment (from LDS): halves 0..7 -> K=k0+hi8.. ; halves 8..15 -> K=k0+16+hi8.., col = t*16+l15
      const f16* brow = bcur + (t * 16 + l15) * 32;
      v8h blo = *(const v8h*)(brow + hi8);
      v8h bhi = *(const v8h*)(brow + 16 + hi8);
      v16h bm;
#pragma unroll
      for (int q = 0; q < 8; ++q) { bm[q] = blo[q]; bm[q + 8] = bhi[q]; }
      acc[t] = __builtin_amdgcn_wmma_f32_16x16x32_f16(
          false, a, false, bm, (short)0, acc[t], false, false);
    }

    async_wait0();        // next chunk resident before anyone passes the barrier
    __syncthreads();      // + all reads of bcur done before it is overwritten
  }

#pragma unroll
  for (int t = 0; t < 4; ++t) {
    const int col = n0 + (t << 4) + l15;
#pragma unroll
    for (int r = 0; r < 8; ++r) {
      const int mrow = m0 + r + hi8;        // C: VGPR r -> row r + 8*(lane>=16)
      C[(size_t)mrow * ldc + col] = (f16)acc[t][r];
    }
  }
}

// =====================================================================
// weight f32 -> f16 with optional row/col zero padding
// =====================================================================
__global__ void cvt_w_f16(const float* __restrict__ src, f16* __restrict__ dst,
                          int rows, int cols, int rows_p, int ldd) {
  size_t i = (size_t)blockIdx.x * blockDim.x + threadIdx.x;
  size_t tot = (size_t)rows_p * ldd;
  if (i >= tot) return;
  int r = (int)(i / ldd), c = (int)(i % ldd);
  float v = (r < rows && c < cols) ? src[(size_t)r * cols + c] : 0.f;
  dst[i] = (f16)v;
}

// =====================================================================
// pointer stats: chain stability / cons, hit scatter
// =====================================================================
__global__ void cons_hit_kernel(const int* __restrict__ ph,
                                float* __restrict__ cons, int* __restrict__ hit) {
  int i = blockIdx.x * blockDim.x + threadIdx.x;       // b*N + n
  if (i >= B_ * N_) return;
  int b = i / N_, n = i % N_;
  int cnt = 0;
#pragma unroll
  for (int t = 0; t < HIST_ - 1; ++t)
    cnt += (ph[(size_t)t * B_ * N_ + i] == ph[(size_t)(t + 1) * B_ * N_ + i]) ? 1 : 0;
  float chain = (float)cnt / (float)(HIST_ - 1);
  float pos_bias = 1.0f - 0.5f * (float)n / (float)(N_ - 1);   // linspace(1,0.5,N)
  cons[i] = chain * pos_bias + 0.1f;
  int last = ph[(size_t)(HIST_ - 1) * B_ * N_ + i];
  hit[b * N_ + last] = 1;                               // benign race (same value)
}

__global__ void __launch_bounds__(256)
diversity_kernel(const int* __restrict__ hit, float* __restrict__ divv) {
  __shared__ float red[256];
  int b = blockIdx.x;
  float acc = 0.f;
  for (int n = threadIdx.x; n < N_; n += 256) acc += (float)hit[b * N_ + n];
  red[threadIdx.x] = acc;
  __syncthreads();
  for (int s = 128; s > 0; s >>= 1) {
    if (threadIdx.x < (unsigned)s) red[threadIdx.x] += red[threadIdx.x + s];
    __syncthreads();
  }
  if (threadIdx.x == 0) divv[b] = red[0] / (float)N_;
}

__global__ void quality_kernel(const int* __restrict__ ph, const float* __restrict__ cons,
                               const float* __restrict__ divv, float* __restrict__ qual) {
  int i = blockIdx.x * blockDim.x + threadIdx.x;
  if (i >= B_ * N_) return;
  int b = i / N_, n = i % N_;
  int last = ph[(size_t)(HIST_ - 1) * B_ * N_ + i];
  float selfp = (last == n) ? 1.0f : 0.0f;
  float q = cons[i] * (0.5f + 0.5f * divv[b]) * (1.0f - 0.3f * selfp);
  qual[i] = fminf(fmaxf(q, 0.1f), 1.0f);
}

// =====================================================================
// ctx = alpha*agg + (1-alpha)*h, then RMS-norm -> f16
// =====================================================================
__global__ void __launch_bounds__(256)
ctx_rms(const float* __restrict__ hsrc, const float* __restrict__ hist,
        const float* __restrict__ cons, const float* __restrict__ rnw,
        f16* __restrict__ ctxn) {
  __shared__ float buf[D_];
  __shared__ float red[256];
  const int tok = blockIdx.x;
  const float c = cons[tok];
  const float wgt = c + 0.1f;
  float ss = 0.f;
  for (int d = threadIdx.x; d < D_; d += 256) {
    float s = 0.f;
#pragma unroll
    for (int t = 0; t < HIST_; ++t)
      s += hist[(size_t)t * (size_t)M_ * D_ + (size_t)tok * D_ + d];
    float agg = (s * (1.f / HIST_)) * wgt;
    float ctx = c * agg + (1.f - c) * hsrc[(size_t)tok * D_ + d];
    buf[d] = ctx;
    ss += ctx * ctx;
  }
  red[threadIdx.x] = ss;
  __syncthreads();
  for (int s2 = 128; s2 > 0; s2 >>= 1) {
    if (threadIdx.x < (unsigned)s2) red[threadIdx.x] += red[threadIdx.x + s2];
    __syncthreads();
  }
  const float inv = rsqrtf(red[0] / (float)D_ + EPS_);
  for (int d = threadIdx.x; d < D_; d += 256)
    ctxn[(size_t)tok * D_ + d] = (f16)(buf[d] * inv * rnw[d]);
}

// generic RMS norm f32 -> f16
__global__ void __launch_bounds__(256)
rms_to_f16(const float* __restrict__ src, const float* __restrict__ w,
           f16* __restrict__ dst) {
  __shared__ float red[256];
  const int tok = blockIdx.x;
  const float* x = src + (size_t)tok * D_;
  float ss = 0.f;
  for (int d = threadIdx.x; d < D_; d += 256) { float v = x[d]; ss += v * v; }
  red[threadIdx.x] = ss;
  __syncthreads();
  for (int s = 128; s > 0; s >>= 1) {
    if (threadIdx.x < (unsigned)s) red[threadIdx.x] += red[threadIdx.x + s];
    __syncthreads();
  }
  const float inv = rsqrtf(red[0] / (float)D_ + EPS_);
  for (int d = threadIdx.x; d < D_; d += 256)
    dst[(size_t)tok * D_ + d] = (f16)(x[d] * inv * w[d]);
}

// =====================================================================
// elementwise residual updates
// =====================================================================
__global__ void add_refl(const float* __restrict__ h, const float* __restrict__ qual,
                         const float* __restrict__ rgate, const f16* __restrict__ refl,
                         float* __restrict__ h1) {
  size_t i = (size_t)blockIdx.x * blockDim.x + threadIdx.x;
  if (i >= (size_t)M_ * D_) return;
  size_t tok = i / D_; int d = (int)(i % D_);
  h1[i] = h[i] + rgate[d] * qual[tok] * (float)refl[i];
}

__global__ void add_gate(const float* __restrict__ h1, const float* __restrict__ gate,
                         const f16* __restrict__ zp, float* __restrict__ h2) {
  size_t i = (size_t)blockIdx.x * blockDim.x + threadIdx.x;
  if (i >= (size_t)M_ * D_) return;
  int d = (int)(i % D_);
  h2[i] = h1[i] + gate[d] * (float)zp[i];
}

__global__ void swiglu_inplace(f16* __restrict__ g, const f16* __restrict__ u) {
  size_t i = (size_t)blockIdx.x * blockDim.x + threadIdx.x;
  if (i >= (size_t)M_ * DFFP_) return;
  float gv = (float)g[i];
  float uv = (float)u[i];
  float s = gv / (1.0f + expf(-gv));          // silu
  g[i] = (f16)(s * uv);
}

__global__ void final_add(const float* __restrict__ h2, const f16* __restrict__ ffn,
                          float* __restrict__ out) {
  size_t i = (size_t)blockIdx.x * blockDim.x + threadIdx.x;
  if (i >= (size_t)M_ * D_) return;
  out[i] = h2[i] + (float)ffn[i];
}

// =====================================================================
// top-8 causal ALiBi attention: one wave32 per (b,h,q) row.
// scores in LDS, 8 successive wave-argmax picks, softmax over 8,
// PV gather, atomic head-averaged pointer-distribution accumulation.
// =====================================================================
__global__ void __launch_bounds__(32)
attn_topk(const f16* __restrict__ qb, const f16* __restrict__ kb,
          const f16* __restrict__ vb, f16* __restrict__ zb,
          float* __restrict__ pfull) {
  __shared__ float sc[N_];
  __shared__ float qv[DH_];
  const int row = blockIdx.x;                 // b*H*N + h*N + qn
  const int qn = row % N_;
  const int h  = (row / N_) % H_;
  const int b  = row / (N_ * H_);
  const int lane = threadIdx.x;
  const size_t tokq = (size_t)b * N_ + qn;

  const f16* qp = qb + tokq * D_ + h * DH_;
  for (int d = lane; d < DH_; d += 32) qv[d] = (float)qp[d];
  __syncthreads();

  const float slope = exp2f(-8.0f * (float)(h + 1) / (float)H_);
  const int nk = qn + 1;                      // causal window
  const f16* kbase = kb + ((size_t)b * N_) * D_ + h * DH_;
  for (int kk = lane; kk < nk; kk += 32) {
    const f16* kp = kbase + (size_t)kk * D_;
    float s = 0.f;
#pragma unroll 8
    for (int d = 0; d < DH_; ++d) s += qv[d] * (float)kp[d];
    sc[kk] = s * 0.125f - slope * (float)(qn - kk);   // 1/sqrt(64) + ALiBi
  }
  __syncthreads();

  const float NEG = -__builtin_inff();
  float topv[TOPK_]; int topi[TOPK_];
  for (int it = 0; it < TOPK_; ++it) {
    float bv = NEG; int bi = 0;
    for (int kk = lane; kk < nk; kk += 32) {
      float v = sc[kk];
      if (v > bv) { bv = v; bi = kk; }
    }
#pragma unroll
    for (int off = 16; off > 0; off >>= 1) {
      float ov = __shfl_xor(bv, off, 32);
      int   oi = __shfl_xor(bi, off, 32);
      if (ov > bv || (ov == bv && oi < bi)) { bv = ov; bi = oi; }
    }
    topv[it] = bv; topi[it] = bi;
    __syncthreads();
    if (lane == 0 && bv > NEG) sc[bi] = NEG;  // exclude picked key
    __syncthreads();
  }

  const float mx = topv[0];                   // first pick is the max
  float pr[TOPK_]; float ssum = 0.f;
#pragma unroll
  for (int i = 0; i < TOPK_; ++i) {
    pr[i] = (topv[i] > NEG) ? expf(topv[i] - mx) : 0.f;
    ssum += pr[i];
  }
  const float inv = 1.f / ssum;

  const f16* vbase = vb + ((size_t)b * N_) * D_ + h * DH_;
  for (int d = lane; d < DH_; d += 32) {
    float acc = 0.f;
#pragma unroll
    for (int i = 0; i < TOPK_; ++i)
      if (pr[i] > 0.f) acc += pr[i] * inv * (float)vbase[(size_t)topi[i] * D_ + d];
    zb[tokq * D_ + h * DH_ + d] = (f16)acc;
  }

  if (lane == 0) {
    float* prow = pfull + tokq * N_;
#pragma unroll
    for (int i = 0; i < TOPK_; ++i)
      if (pr[i] > 0.f) atomicAdd(&prow[topi[i]], pr[i] * inv * (1.f / (float)H_));
  }
}

// argmax/max over head-averaged pointer distribution (first-index ties)
__global__ void __launch_bounds__(32)
pointer_argmax(const float* __restrict__ pfull, float* __restrict__ out_idx,
               float* __restrict__ out_p) {
  const int row = blockIdx.x;                 // token
  const int lane = threadIdx.x;
  const float* pr = pfull + (size_t)row * N_;
  float bv = -1.f; int bi = N_;
  for (int k = lane; k < N_; k += 32) {
    float v = pr[k];
    if (v > bv || (v == bv && k < bi)) { bv = v; bi = k; }
  }
#pragma unroll
  for (int off = 16; off > 0; off >>= 1) {
    float ov = __shfl_xor(bv, off, 32);
    int   oi = __shfl_xor(bi, off, 32);
    if (ov > bv || (ov == bv && oi < bi)) { bv = ov; bi = oi; }
  }
  if (lane == 0) { out_idx[row] = (float)bi; out_p[row] = bv; }
}

// =====================================================================
// driver
// =====================================================================
extern "C" void kernel_launch(void* const* d_in, const int* in_sizes, int n_in,
                              void* d_out, int out_size, void* d_ws, size_t ws_size,
                              hipStream_t stream) {
  (void)in_sizes; (void)n_in; (void)out_size; (void)ws_size;

  const float* h     = (const float*)d_in[0];
  const float* hist  = (const float*)d_in[1];
  const int*   ph    = (const int*)d_in[2];
  const float* w_q   = (const float*)d_in[3];
  const float* w_k   = (const float*)d_in[4];
  const float* w_v   = (const float*)d_in[5];
  const float* w_o   = (const float*)d_in[6];
  const float* gate  = (const float*)d_in[7];
  const float* n1w   = (const float*)d_in[8];
  const float* n2w   = (const float*)d_in[9];
  const float* rgate = (const float*)d_in[10];
  const float* rnw   = (const float*)d_in[11];
  const float* rproj = (const float*)d_in[12];
  const float* wg    = (const float*)d_in[13];
  const float* wu    = (const float*)d_in[14];
  const float* wd    = (const float*)d_in[15];

  // ---- workspace layout ----
  char* p = (char*)d_ws;
  auto carve = [&](size_t bytes) -> void* {
    void* r = (void*)p;
    p += (bytes + 255) & ~(size_t)255;
    return r;
  };
  float* pfull  = (float*)carve((size_t)B_ * N_ * N_ * 4);
  int*   hit    = (int*)  carve((size_t)B_ * N_ * 4);
  float* cons   = (float*)carve((size_t)B_ * N_ * 4);
  float* qual   = (float*)carve((size_t)B_ * N_ * 4);
  float* divv   = (float*)carve((size_t)B_ * 4);
  f16* w_r16 = (f16*)carve((size_t)D_ * D_ * 2);
  f16* w_q16 = (f16*)carve((size_t)D_ * D_ * 2);
  f16* w_k16 = (f16*)carve((size_t)D_ * D_ * 2);
  f16* w_v16 = (f16*)carve((size_t)D_ * D_ * 2);
  f16* w_o16 = (f16*)carve((size_t)D_ * D_ * 2);
  f16* w_g16 = (f16*)carve((size_t)DFFP_ * D_ * 2);
  f16* w_u16 = (f16*)carve((size_t)DFFP_ * D_ * 2);
  f16* w_d16 = (f16*)carve((size_t)D_ * DFFP_ * 2);
  f16* ctxn16 = (f16*)carve((size_t)M_ * D_ * 2);
  f16* refl16 = (f16*)carve((size_t)M_ * D_ * 2);
  float* h1   = (float*)carve((size_t)M_ * D_ * 4);
  f16* x1_16  = (f16*)carve((size_t)M_ * D_ * 2);
  f16* q16    = (f16*)carve((size_t)M_ * D_ * 2);
  f16* k16    = (f16*)carve((size_t)M_ * D_ * 2);
  f16* v16b   = (f16*)carve((size_t)M_ * D_ * 2);
  f16* z16    = (f16*)carve((size_t)M_ * D_ * 2);
  f16* zp16   = (f16*)carve((size_t)M_ * D_ * 2);
  float* h2   = (float*)carve((size_t)M_ * D_ * 4);
  f16* x2_16  = (f16*)carve((size_t)M_ * D_ * 2);
  f16* g16    = (f16*)carve((size_t)M_ * DFFP_ * 2);
  f16* u16    = (f16*)carve((size_t)M_ * DFFP_ * 2);
  f16* ffn16  = (f16*)carve((size_t)M_ * D_ * 2);

  float* out_h   = (float*)d_out;
  float* out_idx = out_h + (size_t)M_ * D_;
  float* out_p   = out_idx + M_;

  auto nb = [](size_t n, int t) { return (unsigned)((n + (size_t)t - 1) / t); };

  // ---- zero the accumulated buffers ----
  hipMemsetAsync(pfull, 0, (size_t)B_ * N_ * N_ * 4, stream);
  hipMemsetAsync(hit, 0, (size_t)B_ * N_ * 4, stream);

  // ---- weight conversion (f32 -> f16, zero padded) ----
  cvt_w_f16<<<nb((size_t)D_ * D_, 256), 256, 0, stream>>>(rproj, w_r16, D_, D_, D_, D_);
  cvt_w_f16<<<nb((size_t)D_ * D_, 256), 256, 0, stream>>>(w_q, w_q16, D_, D_, D_, D_);
  cvt_w_f16<<<nb((size_t)D_ * D_, 256), 256, 0, stream>>>(w_k, w_k16, D_, D_, D_, D_);
  cvt_w_f16<<<nb((size_t)D_ * D_, 256), 256, 0, stream>>>(w_v, w_v16, D_, D_, D_, D_);
  cvt_w_f16<<<nb((size_t)D_ * D_, 256), 256, 0, stream>>>(w_o, w_o16, D_, D_, D_, D_);
  cvt_w_f16<<<nb((size_t)DFFP_ * D_, 256), 256, 0, stream>>>(wg, w_g16, DFF_, D_, DFFP_, D_);
  cvt_w_f16<<<nb((size_t)DFFP_ * D_, 256), 256, 0, stream>>>(wu, w_u16, DFF_, D_, DFFP_, D_);
  cvt_w_f16<<<nb((size_t)D_ * DFFP_, 256), 256, 0, stream>>>(wd, w_d16, D_, DFF_, D_, DFFP_);

  // ---- reflection statistics ----
  cons_hit_kernel<<<nb(B_ * N_, 256), 256, 0, stream>>>(ph, cons, hit);
  diversity_kernel<<<B_, 256, 0, stream>>>(hit, divv);
  quality_kernel<<<nb(B_ * N_, 256), 256, 0, stream>>>(ph, cons, divv, qual);

  // ---- reflection projection ----
  ctx_rms<<<M_, 256, 0, stream>>>(h, hist, cons, rnw, ctxn16);
  dim3 gD(M_ / 64, D_ / 64);                   // (64, 16)
  wmma_gemm_xwT<<<gD, 128, 0, stream>>>(ctxn16, w_r16, refl16, D_, D_, D_, D_);
  add_refl<<<nb((size_t)M_ * D_, 256), 256, 0, stream>>>(h, qual, rgate, refl16, h1);

  // ---- pointer attention block ----
  rms_to_f16<<<M_, 256, 0, stream>>>(h1, n1w, x1_16);
  wmma_gemm_xwT<<<gD, 128, 0, stream>>>(x1_16, w_q16, q16, D_, D_, D_, D_);
  wmma_gemm_xwT<<<gD, 128, 0, stream>>>(x1_16, w_k16, k16, D_, D_, D_, D_);
  wmma_gemm_xwT<<<gD, 128, 0, stream>>>(x1_16, w_v16, v16b, D_, D_, D_, D_);
  attn_topk<<<B_ * H_ * N_, 32, 0, stream>>>(q16, k16, v16b, z16, pfull);
  wmma_gemm_xwT<<<gD, 128, 0, stream>>>(z16, w_o16, zp16, D_, D_, D_, D_);
  add_gate<<<nb((size_t)M_ * D_, 256), 256, 0, stream>>>(h1, gate, zp16, h2);

  // ---- SwiGLU FFN ----
  rms_to_f16<<<M_, 256, 0, stream>>>(h2, n2w, x2_16);
  dim3 gF(M_ / 64, DFFP_ / 64);                // (64, 43)
  wmma_gemm_xwT<<<gF, 128, 0, stream>>>(x2_16, w_g16, g16, D_, D_, D_, DFFP_);
  wmma_gemm_xwT<<<gF, 128, 0, stream>>>(x2_16, w_u16, u16, D_, D_, D_, DFFP_);
  swiglu_inplace<<<nb((size_t)M_ * DFFP_, 256), 256, 0, stream>>>(g16, u16);
  wmma_gemm_xwT<<<gD, 128, 0, stream>>>(g16, w_d16, ffn16, DFFP_, DFFP_, DFFP_, D_);
  final_add<<<nb((size_t)M_ * D_, 256), 256, 0, stream>>>(h2, ffn16, out_h);

  // ---- pointer outputs ----
  pointer_argmax<<<M_, 32, 0, stream>>>(pfull, out_idx, out_p);
}
